// HierarchicalAgent_2723009265993
// MI455X (gfx1250) — matrix operand order repaired and checked by reading, not compiled, not run on hardware
//
#include <hip/hip_runtime.h>
#include <hip/hip_bf16.h>

typedef _Float16 h16;
typedef __attribute__((ext_vector_type(16))) _Float16 v16h;
typedef __attribute__((ext_vector_type(8)))  _Float16 v8h;
typedef __attribute__((ext_vector_type(8)))  float    v8f;

constexpr int BQ   = 32768;  // batch
constexpr int OBSN = 210;
constexpr int ACTN = 200;
constexpr int HIDN = 512;
constexpr int PEMBN = 16;
constexpr int KPAD_IN = 256; // 226 padded to 256
constexpr int LDT  = 40;     // LDS tile stride in halves (32 + 8 pad)
constexpr int TILE_H = 128 * LDT;  // halves per tile buffer

__device__ __constant__ int PH2H[9] = {1, 3, 2, 4, 5, 6, 6, 0, 0};

// ---- CDNA5 async global->LDS path (guarded; falls back to manual staging) --
#if __has_builtin(__builtin_amdgcn_global_load_async_to_lds_b128)
#define HAVE_ASYNC_LDS 1
typedef int v4i_ __attribute__((vector_size(16)));
typedef __attribute__((address_space(1))) v4i_ as1_v4i;
typedef __attribute__((address_space(3))) v4i_ as3_v4i;
__device__ __forceinline__ void async_ld_b128(const h16* g, h16* l) {
  __builtin_amdgcn_global_load_async_to_lds_b128((as1_v4i*)g, (as3_v4i*)l, 0,
                                                 0);
}
#if __has_builtin(__builtin_amdgcn_s_wait_asynccnt)
#define WAIT_ASYNC(n) __builtin_amdgcn_s_wait_asynccnt(n)
#else
#define WAIT_ASYNC(n) asm volatile("s_wait_asynccnt %0" ::"i"(n) : "memory")
#endif
#else
#define HAVE_ASYNC_LDS 0
#endif

// ---------------------------------------------------------------------------
// Prep: build f16 concatenated input [x | phase_emb[pid] | 0-pad] (B x 256)
// and the per-row head-selection table.
// ---------------------------------------------------------------------------
__global__ __launch_bounds__(256) void prep_x16(
    const float* __restrict__ x, const int* __restrict__ phase_ids,
    const float* __restrict__ phase_emb, h16* __restrict__ X16,
    unsigned char* __restrict__ headsel) {
  int row = blockIdx.x;
  int c = threadIdx.x;
  int pid = phase_ids[row];
  float v = 0.0f;
  if (c < OBSN)              v = x[(size_t)row * OBSN + c];
  else if (c < OBSN + PEMBN) v = phase_emb[pid * PEMBN + (c - OBSN)];
  X16[(size_t)row * KPAD_IN + c] = (h16)v;
  if (c == 0) headsel[row] = (unsigned char)PH2H[pid];
}

// ---------------------------------------------------------------------------
// Convert f32 weight (K x N, row-major) to f16 transposed (N x Kpad), zero pad.
// ---------------------------------------------------------------------------
__global__ __launch_bounds__(256) void wconv(
    const float* __restrict__ W, h16* __restrict__ Wt, int K, int N, int Kpad) {
  int idx = blockIdx.x * 256 + threadIdx.x;
  if (idx >= N * Kpad) return;
  int n = idx / Kpad, k = idx - n * Kpad;
  Wt[idx] = (k < K) ? (h16)W[(size_t)k * N + n] : (h16)0.0f;
}

// ---------------------------------------------------------------------------
// LayerNorm over HID=512, one block (256 thr) per row, 2 elems/thread.
// ---------------------------------------------------------------------------
template <bool RELU, bool OUT16>
__global__ __launch_bounds__(256) void ln_kernel(
    const float* __restrict__ in, float* __restrict__ out32,
    h16* __restrict__ out16, const float* __restrict__ g,
    const float* __restrict__ b) {
  __shared__ float red[256];
  int row = blockIdx.x, t = threadIdx.x;
  const float* r = in + (size_t)row * HIDN;
  float x0 = r[t], x1 = r[t + 256];
  red[t] = x0 + x1;
  __syncthreads();
  for (int s = 128; s > 0; s >>= 1) {
    if (t < s) red[t] += red[t + s];
    __syncthreads();
  }
  float mu = red[0] * (1.0f / HIDN);
  __syncthreads();
  red[t] = x0 * x0 + x1 * x1;
  __syncthreads();
  for (int s = 128; s > 0; s >>= 1) {
    if (t < s) red[t] += red[t + s];
    __syncthreads();
  }
  float var = red[0] * (1.0f / HIDN) - mu * mu;
  float rs = rsqrtf(var + 1e-5f);
  float y0 = (x0 - mu) * rs * g[t] + b[t];
  float y1 = (x1 - mu) * rs * g[t + 256] + b[t + 256];
  if (RELU) { y0 = fmaxf(y0, 0.0f); y1 = fmaxf(y1, 0.0f); }
  if (OUT16) {
    out16[(size_t)row * HIDN + t]       = (h16)y0;
    out16[(size_t)row * HIDN + t + 256] = (h16)y1;
  } else {
    out32[(size_t)row * HIDN + t]       = y0;
    out32[(size_t)row * HIDN + t + 256] = y1;
  }
}

// ---------------------------------------------------------------------------
// WMMA GEMM: C(M x N) = A(M x K, f16) * Bt(N x K, f16)^T
// Tile 128x128, BK=32, 256 threads = 8 waves (wave32), each wave 64x32
// via 4x2 grid of v_wmma_f32_16x16x32_f16.
// Async global->LDS double-buffered software pipeline (ASYNCcnt FIFO: after
// issuing the next tile's 4 ops, s_wait_asynccnt 4 guarantees the current
// tile has landed while the next tile's DMA overlaps the WMMA work).
// OOB B-rows are clamped to row N-1: garbage feeds only columns >= N, which
// the epilogue never stores -> staging is branch-free and uniform.
// Epilogue flags: BIAS, RELU, RES (+= f32), S32/S16 stores, MASK (store only
// rows where rowhead[row]==head -> phase-routed heads).
// ---------------------------------------------------------------------------
template <bool BIAS, bool RELU, bool RES, bool S32, bool S16, bool MASK>
__global__ __launch_bounds__(256) void gemm_wmma(
    const h16* __restrict__ A, const h16* __restrict__ Bt,
    const float* __restrict__ bias, const float* __restrict__ resin,
    float* __restrict__ out32, h16* __restrict__ out16,
    const unsigned char* __restrict__ rowhead, int head, int N, int K) {
  __shared__ h16 As[2][TILE_H];
  __shared__ h16 Bs[2][TILE_H];

  const int mbase = blockIdx.x * 128;
  const int nbase = blockIdx.y * 128;
  const int tid = threadIdx.x;
  const int wave = tid >> 5, lane = tid & 31;
  const int lane16 = lane & 15, hivg = lane >> 4;
  const int wm = wave & 1, wn = wave >> 1;

  // staging assignment: each thread owns 2 A-chunks + 2 B-chunks of 16B
  const int lrow = tid >> 2;           // 0..63
  const int lkof = (tid & 3) << 3;     // 0,8,16,24 halves
  const int gn0 = nbase + lrow, gn1 = gn0 + 64;
  const int bn0 = (gn0 < N) ? gn0 : N - 1;   // clamp OOB rows (cols >= N
  const int bn1 = (gn1 < N) ? gn1 : N - 1;   // are never stored)

  const h16* gA0 = A + (size_t)(mbase + lrow) * K + lkof;
  const h16* gA1 = gA0 + (size_t)64 * K;
  const h16* gB0 = Bt + (size_t)bn0 * K + lkof;
  const h16* gB1 = Bt + (size_t)bn1 * K + lkof;
  const int lofs = lrow * LDT + lkof;

  v8f acc[4][2] = {};

#if HAVE_ASYNC_LDS
  // prologue: tile 0 -> buffer 0
  async_ld_b128(gA0, &As[0][lofs]);
  async_ld_b128(gA1, &As[0][lofs + 64 * LDT]);
  async_ld_b128(gB0, &Bs[0][lofs]);
  async_ld_b128(gB1, &Bs[0][lofs + 64 * LDT]);
#endif

  int buf = 0;
  for (int k0 = 0; k0 < K; k0 += 32) {
#if HAVE_ASYNC_LDS
    if (k0 + 32 < K) {  // issue next tile into the other buffer
      int nb = buf ^ 1, kn = k0 + 32;
      async_ld_b128(gA0 + kn, &As[nb][lofs]);
      async_ld_b128(gA1 + kn, &As[nb][lofs + 64 * LDT]);
      async_ld_b128(gB0 + kn, &Bs[nb][lofs]);
      async_ld_b128(gB1 + kn, &Bs[nb][lofs + 64 * LDT]);
      WAIT_ASYNC(4);  // current tile landed; next tile's DMA in flight
    } else {
      WAIT_ASYNC(0);
    }
#else
    // manual staging: issue all 4 global loads, then all 4 LDS stores
    float4 a0 = *(const float4*)(gA0 + k0);
    float4 a1 = *(const float4*)(gA1 + k0);
    float4 b0 = *(const float4*)(gB0 + k0);
    float4 b1 = *(const float4*)(gB1 + k0);
    *(float4*)&As[buf][lofs] = a0;
    *(float4*)&As[buf][lofs + 64 * LDT] = a1;
    *(float4*)&Bs[buf][lofs] = b0;
    *(float4*)&Bs[buf][lofs + 64 * LDT] = b1;
#endif
    __syncthreads();

    // --- fragments per ISA 16-bit A/B VGPR layouts ---
    v16h af[4], bf[2];
#pragma unroll
    for (int i = 0; i < 4; i++) {
      int r = wm * 64 + i * 16 + lane16;
      v8h lo = *(const v8h*)&As[buf][r * LDT + hivg * 8];
      v8h hh = *(const v8h*)&As[buf][r * LDT + 16 + hivg * 8];
#pragma unroll
      for (int t = 0; t < 8; t++) { af[i][t] = lo[t]; af[i][t + 8] = hh[t]; }
    }
#pragma unroll
    for (int j = 0; j < 2; j++) {
      int r = wn * 32 + j * 16 + lane16;
      v8h lo = *(const v8h*)&Bs[buf][r * LDT + hivg * 8];
      v8h hh = *(const v8h*)&Bs[buf][r * LDT + 16 + hivg * 8];
#pragma unroll
      for (int t = 0; t < 8; t++) { bf[j][t] = lo[t]; bf[j][t + 8] = hh[t]; }
    }
#pragma unroll
    for (int i = 0; i < 4; i++)
#pragma unroll
      for (int j = 0; j < 2; j++)
        acc[i][j] = __builtin_amdgcn_wmma_f32_16x16x32_f16(
            false, af[i], false, bf[j], (short)0, acc[i][j], false, false);
    __syncthreads();  // WAR: next iteration's DMA overwrites buf^1
    buf ^= 1;
  }

  // --- epilogue ---
#pragma unroll
  for (int i = 0; i < 4; i++) {
#pragma unroll
    for (int j = 0; j < 2; j++) {
      int col = nbase + wn * 32 + j * 16 + lane16;
      if (col >= N) continue;
#pragma unroll
      for (int v = 0; v < 8; v++) {
        int row = mbase + wm * 64 + i * 16 + hivg * 8 + v;
        if (MASK && rowhead[row] != head) continue;
        float val = acc[i][j][v];
        if (BIAS) val += bias[col];
        if (RELU) val = fmaxf(val, 0.0f);
        if (RES) val += resin[(size_t)row * N + col];
        if (S32) out32[(size_t)row * N + col] = val;
        if (S16) out16[(size_t)row * N + col] = (h16)val;
      }
    }
  }
}

// ---------------------------------------------------------------------------
// Critic final dot: v[row] = c1[row,:] . w + b   (one wave32 per row)
// ---------------------------------------------------------------------------
__global__ __launch_bounds__(256) void critic_dot(
    const h16* __restrict__ c1, const float* __restrict__ w,
    const float* __restrict__ b, float* __restrict__ vout) {
  int wv = threadIdx.x >> 5, lane = threadIdx.x & 31;
  int row = blockIdx.x * 8 + wv;
  float s = 0.0f;
  for (int k = lane; k < HIDN; k += 32)
    s += (float)c1[(size_t)row * HIDN + k] * w[k];
  for (int o = 16; o > 0; o >>= 1) s += __shfl_xor(s, o, 32);
  if (lane == 0) vout[row] = s + b[0];
}

// ---------------------------------------------------------------------------
// Finalize: mask -> log_softmax -> log_prob(action), entropy; pack outputs.
// One wave32 per row, 7 logits per lane.
// ---------------------------------------------------------------------------
__global__ __launch_bounds__(256) void finalize_kernel(
    const float* __restrict__ logits, const float* __restrict__ amask,
    const int* __restrict__ action, const float* __restrict__ vbuf,
    float* __restrict__ out) {
  int wv = threadIdx.x >> 5, lane = threadIdx.x & 31;
  int row = blockIdx.x * 8 + wv;
  float lv[7];
  float mx = -__builtin_inff();
#pragma unroll
  for (int i = 0; i < 7; i++) {
    int c = lane + i * 32;
    float val;
    if (c < ACTN) {
      float l = logits[(size_t)row * ACTN + c];
      val = (amask[(size_t)row * ACTN + c] > 0.5f) ? l : -1.0e8f;
    } else {
      val = -__builtin_inff();
    }
    lv[i] = val;
    mx = fmaxf(mx, val);
  }
  for (int o = 16; o > 0; o >>= 1) mx = fmaxf(mx, __shfl_xor(mx, o, 32));
  float se = 0.0f;
#pragma unroll
  for (int i = 0; i < 7; i++) se += expf(lv[i] - mx);
  for (int o = 16; o > 0; o >>= 1) se += __shfl_xor(se, o, 32);
  float lse = mx + logf(se);
  int act = action[row];
  float lpa = 0.0f, ent = 0.0f;
#pragma unroll
  for (int i = 0; i < 7; i++) {
    int c = lane + i * 32;
    if (c < ACTN) {
      float lp = lv[i] - lse;
      float p = expf(lp);
      ent -= p * lp;
      if (c == act) lpa = lp;
    }
  }
  for (int o = 16; o > 0; o >>= 1) {
    ent += __shfl_xor(ent, o, 32);
    lpa += __shfl_xor(lpa, o, 32);
  }
  if (lane == 0) {
    out[row] = (float)act;
    out[BQ + row] = lpa;
    out[2 * BQ + row] = ent;
    out[3 * BQ + row] = vbuf[row];
  }
}

// ---------------------------------------------------------------------------
extern "C" void kernel_launch(void* const* d_in, const int* in_sizes, int n_in,
                              void* d_out, int out_size, void* d_ws,
                              size_t ws_size, hipStream_t stream) {
  (void)in_sizes; (void)n_in; (void)out_size; (void)ws_size;

  // -------- input mapping (setup_inputs dict order, tuples flattened) -----
  const float* x          = (const float*)d_in[0];
  const float* amask      = (const float*)d_in[1];
  const int*   phase_ids  = (const int*)d_in[2];
  const int*   action     = (const int*)d_in[3];
  const float* phase_emb  = (const float*)d_in[4];
  const float* embed_w    = (const float*)d_in[5];
  const float* embed_b    = (const float*)d_in[6];
  const float* embed_ln_g = (const float*)d_in[7];
  const float* embed_ln_b = (const float*)d_in[8];
  // res_params: 3 x (g, b, w1, b1, w2, b2) -> indices 9..26
  // head_params: 7 x (g, b, w1, b1, w2, b2) -> indices 27..68
  // critic_params: (g, b, w1, b1, w2, b2) -> indices 69..74
  static const int HH[7] = {512, 256, 512, 512, 128, 256, 512};

  // -------- workspace carve -----------------------------------------------
  char* ws = (char*)d_ws;
  size_t off = 0;
  auto carve = [&](size_t bytes) -> void* {
    void* p = ws + off;
    off = (off + bytes + 255) & ~(size_t)255;
    return p;
  };
  h16*   X16    = (h16*)carve((size_t)BQ * KPAD_IN * 2);
  float* hbuf   = (float*)carve((size_t)BQ * HIDN * 4);
  h16*   zbuf   = (h16*)carve((size_t)BQ * HIDN * 2);
  h16*   z1buf  = (h16*)carve((size_t)BQ * HIDN * 2);
  float* logits = (float*)carve((size_t)BQ * ACTN * 4);
  float* vbuf   = (float*)carve((size_t)BQ * 4);
  unsigned char* hsel = (unsigned char*)carve(BQ);
  h16* WembT = (h16*)carve((size_t)HIDN * KPAD_IN * 2);
  h16* WresT[6];
  for (int i = 0; i < 6; i++) WresT[i] = (h16*)carve((size_t)HIDN * HIDN * 2);
  h16* WcritT = (h16*)carve((size_t)HIDN * HIDN * 2);
  h16* Wh1T[7];
  h16* Wh2T[7];
  for (int k = 0; k < 7; k++) {
    Wh1T[k] = (h16*)carve((size_t)HH[k] * HIDN * 2);
    Wh2T[k] = (h16*)carve((size_t)ACTN * HH[k] * 2);
  }

  const int MB = BQ / 128;  // 256 row-tiles

  // -------- prep ----------------------------------------------------------
  prep_x16<<<BQ, 256, 0, stream>>>(x, phase_ids, phase_emb, X16, hsel);
  wconv<<<(HIDN * KPAD_IN + 255) / 256, 256, 0, stream>>>(
      embed_w, WembT, OBSN + PEMBN, HIDN, KPAD_IN);
  for (int i = 0; i < 3; i++) {
    const float* w1 = (const float*)d_in[9 + 6 * i + 2];
    const float* w2 = (const float*)d_in[9 + 6 * i + 4];
    wconv<<<(HIDN * HIDN + 255) / 256, 256, 0, stream>>>(w1, WresT[2 * i],
                                                         HIDN, HIDN, HIDN);
    wconv<<<(HIDN * HIDN + 255) / 256, 256, 0, stream>>>(w2, WresT[2 * i + 1],
                                                         HIDN, HIDN, HIDN);
  }
  wconv<<<(HIDN * HIDN + 255) / 256, 256, 0, stream>>>(
      (const float*)d_in[69 + 2], WcritT, HIDN, HIDN, HIDN);
  for (int k = 0; k < 7; k++) {
    const float* w1 = (const float*)d_in[27 + 6 * k + 2];
    const float* w2 = (const float*)d_in[27 + 6 * k + 4];
    wconv<<<(HH[k] * HIDN + 255) / 256, 256, 0, stream>>>(w1, Wh1T[k], HIDN,
                                                          HH[k], HIDN);
    wconv<<<(ACTN * HH[k] + 255) / 256, 256, 0, stream>>>(w2, Wh2T[k], HH[k],
                                                          ACTN, HH[k]);
  }

  // -------- trunk ---------------------------------------------------------
  gemm_wmma<true, false, false, true, false, false>
      <<<dim3(MB, 4), 256, 0, stream>>>(X16, WembT, embed_b, nullptr, hbuf,
                                        nullptr, nullptr, 0, HIDN, KPAD_IN);
  ln_kernel<true, false><<<BQ, 256, 0, stream>>>(hbuf, hbuf, nullptr,
                                                 embed_ln_g, embed_ln_b);
  for (int i = 0; i < 3; i++) {
    const float* g  = (const float*)d_in[9 + 6 * i + 0];
    const float* bb = (const float*)d_in[9 + 6 * i + 1];
    const float* b1 = (const float*)d_in[9 + 6 * i + 3];
    const float* b2 = (const float*)d_in[9 + 6 * i + 5];
    ln_kernel<false, true><<<BQ, 256, 0, stream>>>(hbuf, nullptr, zbuf, g, bb);
    gemm_wmma<true, true, false, false, true, false>
        <<<dim3(MB, 4), 256, 0, stream>>>(zbuf, WresT[2 * i], b1, nullptr,
                                          nullptr, z1buf, nullptr, 0, HIDN,
                                          HIDN);
    gemm_wmma<true, true, true, true, false, false>
        <<<dim3(MB, 4), 256, 0, stream>>>(z1buf, WresT[2 * i + 1], b2, hbuf,
                                          hbuf, nullptr, nullptr, 0, HIDN,
                                          HIDN);
  }

  // -------- critic --------------------------------------------------------
  {
    const float* cg  = (const float*)d_in[69 + 0];
    const float* cb  = (const float*)d_in[69 + 1];
    const float* cb1 = (const float*)d_in[69 + 3];
    const float* cw2 = (const float*)d_in[69 + 4];
    const float* cb2 = (const float*)d_in[69 + 5];
    ln_kernel<false, true><<<BQ, 256, 0, stream>>>(hbuf, nullptr, zbuf, cg, cb);
    gemm_wmma<true, true, false, false, true, false>
        <<<dim3(MB, 4), 256, 0, stream>>>(zbuf, WcritT, cb1, nullptr, nullptr,
                                          z1buf, nullptr, 0, HIDN, HIDN);
    critic_dot<<<BQ / 8, 256, 0, stream>>>(z1buf, cw2, cb2, vbuf);
  }

  // -------- phase-routed actor heads --------------------------------------
  for (int k = 0; k < 7; k++) {
    const float* g  = (const float*)d_in[27 + 6 * k + 0];
    const float* bb = (const float*)d_in[27 + 6 * k + 1];
    const float* b1 = (const float*)d_in[27 + 6 * k + 3];
    const float* b2 = (const float*)d_in[27 + 6 * k + 5];
    ln_kernel<false, true><<<BQ, 256, 0, stream>>>(hbuf, nullptr, zbuf, g, bb);
    gemm_wmma<true, true, false, false, true, false>
        <<<dim3(MB, HH[k] / 128), 256, 0, stream>>>(zbuf, Wh1T[k], b1, nullptr,
                                                    nullptr, z1buf, nullptr, 0,
                                                    HH[k], HIDN);
    gemm_wmma<true, false, false, true, false, true>
        <<<dim3(MB, 2), 256, 0, stream>>>(z1buf, Wh2T[k], b2, nullptr, logits,
                                          nullptr, hsel, k, ACTN, HH[k]);
  }

  // -------- finalize ------------------------------------------------------
  finalize_kernel<<<BQ / 8, 256, 0, stream>>>(logits, amask, action, vbuf,
                                              (float*)d_out);
}
